// GMSA_90520730731102
// MI455X (gfx1250) — matrix operand order repaired and checked by reading, not compiled, not run on hardware
//
#include <hip/hip_runtime.h>

typedef __attribute__((ext_vector_type(16))) __bf16 v16bf;
typedef __attribute__((ext_vector_type(8)))  __bf16 v8bf;
typedef __attribute__((ext_vector_type(8)))  float  v8f;

#define DEVINL __device__ __forceinline__

// Native f32 -> bf16 convert (hardware v_cvt_pk_bf16_f32 on gfx1250)
DEVINL __bf16 f2bf(float f) { return (__bf16)f; }

// Pack two converted bf16 into one dword for b32 LDS stores
DEVINL unsigned pack2(float f0, float f1) {
  union { __bf16 h[2]; unsigned u; } p;
  p.h[0] = (__bf16)f0;
  p.h[1] = (__bf16)f1;
  return p.u;
}

DEVINL v8f wmma_bf16(v16bf a, v16bf b, v8f c) {
  return __builtin_amdgcn_wmma_f32_16x16x32_bf16(false, a, false, b, (short)0, c,
                                                 false, false);
}

DEVINL v16bf ld_pair(const __bf16* p0, const __bf16* p1) {
  v8bf a = *(const v8bf*)p0;
  v8bf b = *(const v8bf*)p1;
  v16bf r;
#pragma unroll
  for (int i = 0; i < 8; ++i) { r[i] = a[i]; r[i + 8] = b[i]; }
  return r;
}

// A operand (16x32 bf16): row = lane&15; lanes<16 hold K 0-15,
// lanes>=16 hold K 16-31 (contig 16, matches ldB).
DEVINL v16bf ldA(const __bf16* row, int lane) {
  const int k0 = (lane & 16) ? 16 : 0;
  return ld_pair(row + k0, row + k0 + 8);
}
// B operand (32x16 bf16): col = lane&15; lanes<16 K 0-15, lanes>=16 K 16-31 (contig 16).
DEVINL v16bf ldB(const __bf16* p) { return ld_pair(p, p + 8); }

__host__ __device__ constexpr int nwin_c(int si) {
  return 8 * (64 >> si) * (64 >> si);
}

// 32-bit element offset into y (B,192,256,256): scale si, window swi, token t,
// chan ch(0..63). Max 100.7M elements -> fits u32; enables SADDR+voffset (GVS).
DEVINL unsigned src_off(int si, int swi, int t, int ch) {
  const int ws = 4 << si, nw = 64 >> si, wpb = nw * nw;
  const int b = swi / wpb, rem = swi % wpb;
  const int wy = rem / nw, wx = rem % nw;
  const int py = t / ws, px = t % ws;
  const int sh = ws >> 1;
  const int row = (wy * ws + py + sh) & 255;
  const int col = (wx * ws + px + sh) & 255;
  return ((unsigned)(b * 192 + si * 64 + ch) << 16) + (row << 8) + col;
}

// _match_batch index mapping (flat): q-window qw, q-token j -> source window/token
DEVINL void match_map(int nq, int ns, int qw, int j, int& swi, int& t) {
  if (ns == nq)      { swi = qw; t = j; }
  else if (ns < nq)  { const int f = nq / ns; swi = qw / f; t = j * f + (qw & (f - 1)); }
  else               { const int f = ns / nq; swi = qw * f + (j & (f - 1)); t = j / f; }
}

// ---------------------------------------------------------------------------
// 1x1 conv as WMMA GEMM: out[b,o,hw] = sum_c in[b,c,hw]*w[o,c], + per-o scale/bias
// Cin = 96 (3 bf16 K-tiles). One 16-pixel x 16-channel tile per wave.
// ---------------------------------------------------------------------------
__global__ __launch_bounds__(256) void proj_kernel(
    const float* __restrict__ in, const float* __restrict__ w,
    const float* __restrict__ bias, const float* __restrict__ gamma,
    const float* __restrict__ beta, const float* __restrict__ mean,
    const float* __restrict__ var, int cout, int use_bn,
    float* __restrict__ out) {
  const int lane = threadIdx.x & 31;
  const int wid = (blockIdx.x * 256 + threadIdx.x) >> 5;
  const int ntiles = cout >> 4;
  const int mt = wid / ntiles;
  const int nt = wid % ntiles;
  const int b = mt >> 12;              // 65536/16 = 4096 pixel tiles per image
  const int hw0 = (mt & 4095) << 4;
  const int m = lane & 15;             // pixel within tile (A row)
  const int o = nt * 16 + m;           // output channel (B col / D col)
  const float* xb = in + (long long)b * 96 * 65536;

  v8f c = {};
#pragma unroll
  for (int ktile = 0; ktile < 3; ++ktile) {
    v16bf A;
    const unsigned ka = ktile * 32 + ((lane & 16) ? 8 : 0);
    const unsigned abase = hw0 + m;
#pragma unroll
    for (int j = 0; j < 8; ++j)
      A[j] = f2bf(xb[((ka + j) << 16) + abase]);
#pragma unroll
    for (int j = 0; j < 8; ++j)
      A[8 + j] = f2bf(xb[((ka + 16 + j) << 16) + abase]);
    v16bf B;
    const int kb = ktile * 32 + ((lane & 16) ? 8 : 0);
    const float4* wr4 = (const float4*)(w + o * 96 + kb);  // 16B aligned
#pragma unroll
    for (int q4 = 0; q4 < 4; ++q4) {
      const float4 f = wr4[q4 + (q4 & 2)];
      B[q4 * 4 + 0] = f2bf(f.x);
      B[q4 * 4 + 1] = f2bf(f.y);
      B[q4 * 4 + 2] = f2bf(f.z);
      B[q4 * 4 + 3] = f2bf(f.w);
    }
    c = wmma_bf16(A, B, c);
  }

  float s = 1.0f, t;
  if (use_bn) {
    const float inv = gamma[o] * rsqrtf(var[o] + 1e-5f);
    s = inv;
    t = bias[o] * inv + beta[o] - mean[o] * inv;
  } else {
    t = bias[o];
  }
  // rows r map to 8 contiguous pixels -> two b128 stores
  float* ob = out + ((long long)b * cout + o) * 65536 + hw0 + ((lane & 16) ? 8 : 0);
  float4 s0, s1;
  s0.x = c[0] * s + t; s0.y = c[1] * s + t; s0.z = c[2] * s + t; s0.w = c[3] * s + t;
  s1.x = c[4] * s + t; s1.y = c[5] * s + t; s1.z = c[6] * s + t; s1.w = c[7] * s + t;
  *(float4*)(ob) = s0;
  *(float4*)(ob + 4) = s1;
}

// ---------------------------------------------------------------------------
// Windowed cross-scale attention. One block per q-window; one wave per 16-row
// strip (S=256: 8 waves x 2 strips). q/k/v^T staged to LDS bf16 (packed b32
// stores); scores kept in VGPRs; softmax via half-wave shuffles; P relaid
// C->A layout via a wave-private LDS tile (s_wait_dscnt, LDS in-order/wave).
// ---------------------------------------------------------------------------
template <int S, int QSI, int KSI, int VSI, int COMBO>
__global__ __launch_bounds__(32 * ((S / 16 < 8) ? S / 16 : 8))
void attn_kernel(const float* __restrict__ y, float* __restrict__ y2) {
  constexpr int CT = S / 16;
  constexpr int NWAVES = (CT < 8) ? CT : 8;
  constexpr int BLK = 32 * NWAVES;
  constexpr int SP = (S < 32) ? 32 : S;    // K-dim of P@V padded to 32
  constexpr int NQ = nwin_c(QSI);
  constexpr int NK = nwin_c(KSI);
  constexpr int NV = nwin_c(VSI);
  constexpr int WS = 4 << QSI;
  constexpr int NWG = 64 >> QSI;

  __shared__ __align__(16) __bf16 qbuf[S * 32];            // [token][chan]
  __shared__ __align__(16) __bf16 kbuf[S * 32];            // [token][chan]
  __shared__ __align__(16) __bf16 vbuf[32 * SP];           // [chan][token] (transposed)
  __shared__ __align__(16) __bf16 pwbuf[NWAVES * 16 * 32]; // per-wave P tile

  const int tid = threadIdx.x;
  const int lane = tid & 31;
  const int wid = tid >> 5;
  const int qw = blockIdx.x;

  // ---- stage q/k (window + roll + _match_batch gather), packed bf16x2 ----
  for (int idx = tid; idx < S * 16; idx += BLK) {
    const int c0 = (idx & 15) * 2;       // channel pair
    const int j = idx >> 4;              // token
    *(unsigned*)&qbuf[j * 32 + c0] =
        pack2(y[src_off(QSI, qw, j, c0)], y[src_off(QSI, qw, j, c0 + 1)]);
    int swi, t;
    match_map(NQ, NK, qw, j, swi, t);
    *(unsigned*)&kbuf[j * 32 + c0] =
        pack2(y[src_off(KSI, swi, t, c0)], y[src_off(KSI, swi, t, c0 + 1)]);
  }
  // ---- stage v transposed: pack along token pairs ----
  for (int idx = tid; idx < S * 16; idx += BLK) {
    const int c = idx / (S / 2);         // channel
    const int j0 = (idx % (S / 2)) * 2;  // token pair
    int swi0, t0, swi1, t1;
    match_map(NQ, NV, qw, j0, swi0, t0);
    match_map(NQ, NV, qw, j0 + 1, swi1, t1);
    *(unsigned*)&vbuf[c * SP + j0] =
        pack2(y[src_off(VSI, swi0, t0, 32 + c)], y[src_off(VSI, swi1, t1, 32 + c)]);
  }
  if constexpr (SP > S) {
    for (int idx = tid; idx < 16 * (SP - S); idx += BLK) {
      const int c = idx / ((SP - S) / 2);
      const int j0 = S + (idx % ((SP - S) / 2)) * 2;
      *(unsigned*)&vbuf[c * SP + j0] = 0u;
    }
  }
  __syncthreads();

  constexpr int WPB = NWG * NWG;
  const int bimg = qw / WPB;
  const int wrem = qw % WPB;
  const int wy = wrem / NWG;
  const int wx = wrem % NWG;

  const int mA = lane & 15;
  const int mrow = (lane & 16) ? 8 : 0;
  const int coll = lane & 15;
  __bf16* pw = pwbuf + wid * (16 * 32);
  float* outb = y2 + (long long)bimg * 96 * 65536;

  for (int rt = wid; rt < CT; rt += NWAVES) {
    // ---- scores: Q(16x32) x K^T -> 16 x S strip in VGPRs ----
    v16bf QA = ldA(qbuf + (rt * 16 + mA) * 32, lane);
    v8f acc[CT];
#pragma unroll
    for (int ct = 0; ct < CT; ++ct) {
      const int j = ct * 16 + coll;                 // key token (B col)
      const int k0c = (lane & 16) ? 16 : 0;         // channel base (B K-dim)
      v16bf KB = ldB(kbuf + j * 32 + k0c);
      v8f z = {};
      acc[ct] = wmma_bf16(QA, KB, z);
    }

    // ---- row softmax: reduce over register tiles + 16 lanes of the half ----
    float rinv[8];
#pragma unroll
    for (int r = 0; r < 8; ++r) {
      float mx = acc[0][r];
#pragma unroll
      for (int ct = 1; ct < CT; ++ct) mx = fmaxf(mx, acc[ct][r]);
      mx = fmaxf(mx, __shfl_xor(mx, 1));
      mx = fmaxf(mx, __shfl_xor(mx, 2));
      mx = fmaxf(mx, __shfl_xor(mx, 4));
      mx = fmaxf(mx, __shfl_xor(mx, 8));
      float s = 0.0f;
#pragma unroll
      for (int ct = 0; ct < CT; ++ct) {
        const float e = __expf(acc[ct][r] - mx);
        acc[ct][r] = e;
        s += e;
      }
      s += __shfl_xor(s, 1);
      s += __shfl_xor(s, 2);
      s += __shfl_xor(s, 4);
      s += __shfl_xor(s, 8);
      rinv[r] = 1.0f / s;
    }

    // ---- P @ V: relayout P through wave-private LDS tile, 2 WMMA per K-tile ----
    v8f o0 = {}, o1 = {};
#pragma unroll
    for (int kt = 0; kt < SP / 32; ++kt) {
#pragma unroll
      for (int half = 0; half < 2; ++half) {
        const int ct = kt * 2 + half;
#pragma unroll
        for (int r = 0; r < 8; ++r) {
          const float v = (ct < CT) ? acc[ct][r] * rinv[r] : 0.0f;
          pw[(mrow + r) * 32 + half * 16 + coll] = f2bf(v);
        }
      }
      asm volatile("s_wait_dscnt 0" ::: "memory");  // wave-local LDS RAW
      v16bf PA = ldA(pw + mA * 32, lane);
      const int jt = kt * 32 + ((lane & 16) ? 16 : 0);
      v16bf VB0 = ldB(vbuf + coll * SP + jt);
      v16bf VB1 = ldB(vbuf + (16 + coll) * SP + jt);
      o0 = wmma_bf16(PA, VB0, o0);
      o1 = wmma_bf16(PA, VB1, o1);
    }

    // ---- un-window + roll-back scatter into y2 channels [COMBO*32, +32) ----
#pragma unroll
    for (int r = 0; r < 8; ++r) {
      const int t = rt * 16 + mrow + r;
      const int py = t / WS, px = t % WS;
      const int row = (wy * WS + py + WS / 2) & 255;
      const int col = (wx * WS + px + WS / 2) & 255;
      const unsigned poff = (unsigned)(row << 8) + col;
      outb[((unsigned)(COMBO * 32 + coll) << 16) + poff] = o0[r];
      outb[((unsigned)(COMBO * 32 + 16 + coll) << 16) + poff] = o1[r];
    }
  }
}

// ---------------------------------------------------------------------------
extern "C" void kernel_launch(void* const* d_in, const int* in_sizes, int n_in,
                              void* d_out, int out_size, void* d_ws, size_t ws_size,
                              hipStream_t stream) {
  (void)in_sizes; (void)n_in; (void)out_size; (void)ws_size;
  const float* x     = (const float*)d_in[0];
  const float* w_in  = (const float*)d_in[1];
  const float* b_in  = (const float*)d_in[2];
  const float* bn_g  = (const float*)d_in[3];
  const float* bn_b  = (const float*)d_in[4];
  const float* bn_m  = (const float*)d_in[5];
  const float* bn_v  = (const float*)d_in[6];
  const float* w_out = (const float*)d_in[7];
  const float* b_out = (const float*)d_in[8];
  float* out = (float*)d_out;
  float* y   = (float*)d_ws;                     // (8,192,256,256) f32 = 384 MiB
  float* y2  = y + (size_t)8 * 192 * 65536;      // (8, 96,256,256) f32 = 192 MiB

  // y = BN(x @ w_in^T + b_in):  8*4096 pixel-tiles * 12 o-tiles / 8 waves
  proj_kernel<<<49152, 256, 0, stream>>>(x, w_in, b_in, bn_g, bn_b, bn_m, bn_v,
                                         192, 1, y);
  // combo 0: q=ws4(S=16), k=ws8, v=ws16   -> y2 channels [0,32)
  attn_kernel<16, 0, 1, 2, 0><<<32768, 32, 0, stream>>>(y, y2);
  // combo 1: q=ws8(S=64), k=ws16, v=ws4   -> y2 channels [32,64)
  attn_kernel<64, 1, 2, 0, 1><<<8192, 128, 0, stream>>>(y, y2);
  // combo 2: q=ws16(S=256), k=ws4, v=ws8  -> y2 channels [64,96)
  attn_kernel<256, 2, 0, 1, 2><<<2048, 256, 0, stream>>>(y, y2);
  // out = y2 @ w_out^T + b_out
  proj_kernel<<<24576, 256, 0, stream>>>(y2, w_out, b_out, nullptr, nullptr,
                                         nullptr, nullptr, 96, 0, out);
}
